// PointTransformerLayer_12309376270549
// MI455X (gfx1250) — matrix operand order repaired and checked by reading, not compiled, not run on hardware
//
#include <hip/hip_runtime.h>
#include <hip/hip_fp16.h>

typedef __attribute__((ext_vector_type(16))) _Float16 v16h;
typedef __attribute__((ext_vector_type(8)))  float    v8f;

#define NPTS 1024
#define DIMV 64
#define AH   256
#define WAVES 4
#define BLK  (WAVES*32)
#define LOG2E 1.44269504088896340736f

union H16U { v16h h; unsigned u[8]; };

static __device__ __forceinline__ v8f wmma_f16(v16h a, v16h b, v8f c) {
  // D = A(16x32 f16) * B(32x16 f16) + C(16x16 f32)
  return __builtin_amdgcn_wmma_f32_16x16x32_f16(false, a, false, b, (short)0, c, false, false);
}

// A fragment (16x32 f16) from LDS, row-major [16][strideHalves].
// ISA layout: lanes 0-15 -> M=lane, K = kOff+{0..7,16..23}; lanes 16-31 -> K = kOff+{8..15,24..31}
static __device__ __forceinline__ v16h load_afrag(const _Float16* base, int strideHalves, int kOff, int lane) {
  const unsigned* p = (const unsigned*)base;
  int row = lane & 15;
  int kb  = (lane & 16) ? 8 : 0;
  int rb  = row * (strideHalves >> 1);
  H16U r;
#pragma unroll
  for (int v = 0; v < 4; ++v) r.u[v]     = p[rb + ((kOff + kb + 2*v) >> 1)];
#pragma unroll
  for (int v = 0; v < 4; ++v) r.u[4 + v] = p[rb + ((kOff + 16 + kb + 2*v) >> 1)];
  return r.h;
}

// B fragment (32x16 f16) from global f16 weights stored transposed wt[n][K] (K = strideHalves).
// ISA layout: lane = N (0-15), lanes 0-15 hold K=kOff+{0..7,16..23}, lanes 16-31 hold K=kOff+{8..15,24..31}.
static __device__ __forceinline__ v16h load_bfrag(const _Float16* wt, int n, int strideHalves, int kOff, int lane) {
  int kb = (lane & 16) ? 8 : 0;
  const unsigned* p = (const unsigned*)(wt + n * strideHalves + kOff + kb);
  H16U r;
#pragma unroll
  for (int v = 0; v < 8; ++v) r.u[v] = p[(v & 3) + ((v >> 2) << 3)];
  return r.h;
}

// ---------- prep kernel 1: qkv = x @ W_qkv (f32), tiny ----------
__global__ void pt_qkv_12309(const float* __restrict__ x, const float* __restrict__ Wqkv,
                             float* __restrict__ qkv) {
  __shared__ float xr[DIMV];
  int n = blockIdx.x;
  int c = threadIdx.x;                 // 0..191
  if (c < DIMV) xr[c] = x[n * DIMV + c];
  __syncthreads();
  float acc = 0.f;
#pragma unroll 8
  for (int k = 0; k < DIMV; ++k) acc += xr[k] * Wqkv[k * 192 + c];
  qkv[n * 192 + c] = acc;
}

// ---------- prep kernel 2: transpose + fp32->f16 weights ----------
// W_attn2 is pre-scaled by log2(e): sim feeds only the softmax, so running the
// online softmax in base-2 (v_exp_f32 is natively 2^x) is exact and saves a
// v_mul per exponential.
__global__ void pt_prep_12309(const float* __restrict__ Wpos2, const float* __restrict__ Wattn1,
                              const float* __restrict__ Wattn2,
                              _Float16* __restrict__ wtPos2, _Float16* __restrict__ wtA1,
                              _Float16* __restrict__ wtA2) {
  int t = blockIdx.x * blockDim.x + threadIdx.x;     // 0..16383
  if (t < DIMV * DIMV) { int n = t >> 6, k = t & 63;  wtPos2[n * DIMV + k] = (_Float16)Wpos2[k * DIMV + n]; }
  if (t < AH * DIMV)   { int n = t >> 6, k = t & 63;  wtA1[n * DIMV + k]   = (_Float16)Wattn1[k * AH + n]; }
  if (t < DIMV * AH)   { int n = t >> 8, k = t & 255; wtA2[n * AH + k]     = (_Float16)(Wattn2[k * DIMV + n] * LOG2E); }
}

// ---------- main fused kernel: one block per query i ----------
// waves_per_eu(2): cap VGPRs at ~512 so 2 waves/SIMD co-reside. The anti-LICM
// asm barrier below keeps the big weight matrices' B-fragments as in-loop
// WGP$-hitting loads instead of 576 hoisted VGPRs (round 1) or scratch spills
// (round 2).
__global__ __launch_bounds__(BLK) __attribute__((amdgpu_waves_per_eu(2)))
void pt_main_12309(
    const float* __restrict__ pos,        // 1024*3
    const float* __restrict__ qkv,        // 1024*192 (q|k|v)
    const float* __restrict__ Wpos1,      // 3*64
    const float* __restrict__ bpos1,      // 64
    const _Float16* __restrict__ wtPos2,  // [n=64][k=64]
    const float* __restrict__ bpos2,      // 64
    const _Float16* __restrict__ wtA1,    // [n=256][k=64]
    const float* __restrict__ bA1,        // 256
    const _Float16* __restrict__ wtA2,    // [n=64][k=256], pre-scaled by log2(e)
    const float* __restrict__ bA2,        // 64
    float* __restrict__ out) {            // 1024*64
  __shared__ __align__(16) _Float16 sA[WAVES][16 * DIMV];   // h1 then s (per-wave)
  __shared__ __align__(16) _Float16 sH2[WAVES][16 * AH];    // h2 (per-wave)
  __shared__ float sM[2 * WAVES][DIMV];
  __shared__ float sL[2 * WAVES][DIMV];
  __shared__ float sAc[2 * WAVES][DIMV];

  const int i    = blockIdx.x;
  const int tid  = threadIdx.x;
  const int wave = tid >> 5;
  const int lane = tid & 31;
  const int col  = lane & 15;
  const int hi   = (lane >> 4) & 1;

  // loop-invariant per-lane data
  const float pix = pos[i * 3 + 0], piy = pos[i * 3 + 1], piz = pos[i * 3 + 2];
  float qv[4];
#pragma unroll
  for (int nt = 0; nt < 4; ++nt) qv[nt] = qkv[i * 192 + nt * 16 + col];

  const int h0 = lane, h1i = lane + 32;  // this lane's two hidden channels of pos-MLP layer 1
  const float w10a = Wpos1[h0],  w11a = Wpos1[64 + h0],  w12a = Wpos1[128 + h0],  b1a = bpos1[h0];
  const float w10b = Wpos1[h1i], w11b = Wpos1[64 + h1i], w12b = Wpos1[128 + h1i], b1b = bpos1[h1i];

  float bias2[4], biasS[4];
#pragma unroll
  for (int nt = 0; nt < 4; ++nt) {
    bias2[nt] = bpos2[nt * 16 + col];
    biasS[nt] = bA2[nt * 16 + col] * LOG2E;   // base-2 softmax domain
  }

  float m[4], l[4], ac[4];
#pragma unroll
  for (int nt = 0; nt < 4; ++nt) { m[nt] = -1e30f; l[nt] = 0.f; ac[nt] = 0.f; }

  _Float16* myA  = &sA[wave][0];
  _Float16* myH2 = &sH2[wave][0];

  // Local copies of the big weight pointers; an asm barrier inside the loop
  // makes them loop-variant so their fragment loads are NOT hoisted/spilled.
  const _Float16* wA1 = wtA1;
  const _Float16* wA2 = wtA2;

  for (int jt = wave; jt < NPTS / 16; jt += WAVES) {
    const int jbase = jt * 16;

    // defeat LICM on the weight streams (they hit WGP$/L2 every iteration)
    asm volatile("" : "+s"(wA1), "+s"(wA2));

    // prefetch next tile's k/v rows (global_prefetch_b8); qkv streams once per block
    if (jt + WAVES < NPTS / 16) {
      __builtin_prefetch(qkv + (jbase + WAVES * 16) * 192 + lane * 24, 0, 1);
      __builtin_prefetch(pos + (jbase + WAVES * 16) * 3, 0, 1);
    }

    // ---- stage 1: h1 = relu(relpos @ Wpos1 + b1) -> sA (16x64 f16), VALU (K=3 is tiny)
#pragma unroll 4
    for (int r = 0; r < 16; ++r) {
      int j = jbase + r;
      float rx = pix - pos[j * 3 + 0];
      float ry = piy - pos[j * 3 + 1];
      float rz = piz - pos[j * 3 + 2];
      float va = fmaxf(b1a + rx * w10a + ry * w11a + rz * w12a, 0.f);
      float vb = fmaxf(b1b + rx * w10b + ry * w11b + rz * w12b, 0.f);
      myA[r * DIMV + h0]  = (_Float16)va;
      myA[r * DIMV + h1i] = (_Float16)vb;
    }
    __builtin_amdgcn_wave_barrier();

    // ---- GEMM1: e = h1 @ W_pos2 + b_pos2 (16x64, K=64 -> 8 WMMAs)
    // wtPos2 fragments are small (64 VGPRs) -> fine for the compiler to hoist.
    v16h a0 = load_afrag(myA, DIMV, 0, lane);
    v16h a1 = load_afrag(myA, DIMV, 32, lane);
    v8f e[4];
#pragma unroll
    for (int nt = 0; nt < 4; ++nt) {
      int n = nt * 16 + col;
      v8f c;
#pragma unroll
      for (int r = 0; r < 8; ++r) c[r] = bias2[nt];
      c = wmma_f16(a0, load_bfrag(wtPos2, n, DIMV, 0, lane), c);
      c = wmma_f16(a1, load_bfrag(wtPos2, n, DIMV, 32, lane), c);
      e[nt] = c;
    }
    __builtin_amdgcn_wave_barrier();

    // ---- s = q_i - k_j + e -> sA (overwrite), f16
#pragma unroll
    for (int nt = 0; nt < 4; ++nt) {
      int d = nt * 16 + col;
#pragma unroll
      for (int r = 0; r < 8; ++r) {
        int row = r + hi * 8;
        int j = jbase + row;
        float sv = qv[nt] - qkv[j * 192 + 64 + d] + e[nt][r];
        myA[row * DIMV + d] = (_Float16)sv;
      }
    }
    __builtin_amdgcn_wave_barrier();

    // ---- GEMM2: h2 = relu(s @ W_attn1 + b1) -> sH2 (16x256, 32 WMMAs)
    a0 = load_afrag(myA, DIMV, 0, lane);
    a1 = load_afrag(myA, DIMV, 32, lane);
#pragma unroll
    for (int nt2 = 0; nt2 < 16; ++nt2) {
      int n = nt2 * 16 + col;
      float bb = bA1[n];
      v8f c;
#pragma unroll
      for (int r = 0; r < 8; ++r) c[r] = bb;
      c = wmma_f16(a0, load_bfrag(wA1, n, DIMV, 0, lane), c);
      c = wmma_f16(a1, load_bfrag(wA1, n, DIMV, 32, lane), c);
#pragma unroll
      for (int r = 0; r < 8; ++r) {
        int row = r + hi * 8;
        myH2[row * AH + n] = (_Float16)fmaxf(c[r], 0.f);
      }
    }
    __builtin_amdgcn_wave_barrier();

    // ---- GEMM3: sim2 = log2e*(h2 @ W_attn2 + b2) (16x64, K=256 -> 32 WMMAs)
    v16h af[8];
#pragma unroll
    for (int ks = 0; ks < 8; ++ks) af[ks] = load_afrag(myH2, AH, ks * 32, lane);
#pragma unroll
    for (int nt = 0; nt < 4; ++nt) {
      int d = nt * 16 + col;
      v8f c;
#pragma unroll
      for (int r = 0; r < 8; ++r) c[r] = biasS[nt];
#pragma unroll
      for (int ks = 0; ks < 8; ++ks)
        c = wmma_f16(af[ks], load_bfrag(wA2, d, AH, ks * 32, lane), c);

      // ---- online softmax over rows (j) for this column d, base-2 flash-style
      float tmax = c[0];
#pragma unroll
      for (int r = 1; r < 8; ++r) tmax = fmaxf(tmax, c[r]);
      float mnew = fmaxf(m[nt], tmax);
      float sc = exp2f(m[nt] - mnew);
      l[nt] *= sc; ac[nt] *= sc;
#pragma unroll
      for (int r = 0; r < 8; ++r) {
        int row = r + hi * 8;
        int j = jbase + row;
        float ve = qkv[j * 192 + 128 + d] + e[nt][r];   // v_j + rel_pos_emb
        float wj = exp2f(c[r] - mnew);
        l[nt] += wj;
        ac[nt] += wj * ve;
      }
      m[nt] = mnew;
    }
  }

  // ---- combine 4 waves x 2 lane-halves partial softmax states per column
  int slot = wave * 2 + hi;
#pragma unroll
  for (int nt = 0; nt < 4; ++nt) {
    int d = nt * 16 + col;
    sM[slot][d] = m[nt]; sL[slot][d] = l[nt]; sAc[slot][d] = ac[nt];
  }
  __syncthreads();
  if (tid < DIMV) {
    float M = -1e30f;
#pragma unroll
    for (int s = 0; s < 2 * WAVES; ++s) M = fmaxf(M, sM[s][tid]);
    float L = 0.f, A = 0.f;
#pragma unroll
    for (int s = 0; s < 2 * WAVES; ++s) {
      float sc = exp2f(sM[s][tid] - M);
      L += sL[s][tid] * sc;
      A += sAc[s][tid] * sc;
    }
    out[i * DIMV + tid] = A / L;
  }
}

extern "C" void kernel_launch(void* const* d_in, const int* in_sizes, int n_in,
                              void* d_out, int out_size, void* d_ws, size_t ws_size,
                              hipStream_t stream) {
  const float* x      = (const float*)d_in[0];
  const float* pos    = (const float*)d_in[1];
  const float* Wqkv   = (const float*)d_in[2];
  const float* Wpos1  = (const float*)d_in[3];
  const float* bpos1  = (const float*)d_in[4];
  const float* Wpos2  = (const float*)d_in[5];
  const float* bpos2  = (const float*)d_in[6];
  const float* Wattn1 = (const float*)d_in[7];
  const float* bA1    = (const float*)d_in[8];
  const float* Wattn2 = (const float*)d_in[9];
  const float* bA2    = (const float*)d_in[10];
  float* out = (float*)d_out;

  char* ws = (char*)d_ws;
  float*    qkv    = (float*)ws;                  // 1024*192*4 = 786432 B
  _Float16* wtPos2 = (_Float16*)(ws + 786432);    //  8192 B
  _Float16* wtA1   = (_Float16*)(ws + 794624);    // 32768 B
  _Float16* wtA2   = (_Float16*)(ws + 827392);    // 32768 B

  pt_qkv_12309<<<dim3(NPTS), dim3(192), 0, stream>>>(x, Wqkv, qkv);
  pt_prep_12309<<<dim3(64), dim3(256), 0, stream>>>(Wpos2, Wattn1, Wattn2, wtPos2, wtA1, wtA2);
  pt_main_12309<<<dim3(NPTS), dim3(BLK), 0, stream>>>(pos, qkv, Wpos1, bpos1, wtPos2, bpos2,
                                                      wtA1, bA1, wtA2, bA2, out);
}
